// CgpHmmCell_68341519614147
// MI455X (gfx1250) — 8-bit matrix operand order repaired and checked by reading, compiled, not run on hardware
//
#include <hip/hip_runtime.h>

#define B_SZ   128
#define T_LEN  8192
#define M_DIM  64
#define S_DIM  512
#define F_EPS  1e-16f
#define A_SCALE     64.0f
#define INV_A_SCALE (1.0f / 64.0f)

typedef int   v8i  __attribute__((ext_vector_type(8)));
typedef int   v16i __attribute__((ext_vector_type(16)));
typedef float v8f  __attribute__((ext_vector_type(8)));

__device__ inline unsigned int f32_to_fp8_e4m3(float f) {
  union { float f; unsigned u; } v; v.f = f;
  unsigned s = (v.u >> 24) & 0x80u;
  unsigned u = v.u & 0x7fffffffu;
  if (u >= 0x43e00000u) return s | 0x7Eu;
  int e8 = (int)(u >> 23) - 127 + 7;
  unsigned mant = u & 0x7fffffu;
  if (e8 <= 0) {
    float a = v.f < 0.f ? -v.f : v.f;
    int q = (int)(a * 512.0f + 0.5f);
    if (q > 7) q = 7;
    return s | (unsigned)q;
  }
  mant += 1u << 19;
  if (mant >> 23) { mant = 0; e8 += 1; if (e8 >= 16) return s | 0x7Eu; }
  unsigned r = s | ((unsigned)e8 << 3) | (mant >> 20);
  if ((r & 0x7fu) == 0x7fu) r = s | 0x7Eu;
  return r;
}

__global__ void k_conv_A(const float* __restrict__ A, unsigned char* __restrict__ A8t) {
  int idx = blockIdx.x * blockDim.x + threadIdx.x;
  if (idx >= S_DIM * S_DIM) return;
  int n = idx >> 9, k = idx & (S_DIM - 1);
  A8t[idx] = (unsigned char)f32_to_fp8_e4m3(A[k * S_DIM + n] * A_SCALE);
}

__global__ void k_logB(const float* __restrict__ Bm, float* __restrict__ logB) {
  int idx = blockIdx.x * blockDim.x + threadIdx.x;
  if (idx >= M_DIM * S_DIM) return;
  logB[idx] = __logf(Bm[idx] + F_EPS);
}

__global__ void k_sym(const float* __restrict__ x, unsigned char* __restrict__ sym) {
  int idx = blockIdx.x * blockDim.x + threadIdx.x;
  if (idx >= B_SZ * T_LEN) return;
  const float4* p = (const float4*)(x + (size_t)idx * M_DIM);
  int best = 0;
  #pragma unroll
  for (int j = 0; j < M_DIM / 4; ++j) {
    float4 v = p[j];
    if (v.x > 0.5f) best = 4 * j + 0;
    if (v.y > 0.5f) best = 4 * j + 1;
    if (v.z > 0.5f) best = 4 * j + 2;
    if (v.w > 0.5f) best = 4 * j + 3;
  }
  sym[idx] = (unsigned char)best;
}

#define LDS_TOTAL 304256

__global__ void __launch_bounds__(256, 1)
k_forward(const unsigned char* __restrict__ A8t,
          const float* __restrict__ logB,
          const unsigned char* __restrict__ sym,
          float* __restrict__ out_alpha,
          float* __restrict__ out_ll)
{
  extern __shared__ unsigned char smem[];
  unsigned char* A_lds = smem;
  float*         alpha = (float*)(smem + 262144);
  unsigned char* P_lds = smem + 294912;
  float*         red   = (float*)(smem + 303104);
  float*         mrow  = (float*)(smem + 304128);
  int*           symb  = (int*)(smem + 304192);

  const int tid  = threadIdx.x;
  const int lane = tid & 31;
  const int wave = tid >> 5;
  const int b0   = blockIdx.x * 16;

  {
    const uint4* g = (const uint4*)A8t;
    uint4*       l = (uint4*)A_lds;
    for (int i = tid; i < (S_DIM * S_DIM) / 16; i += 256) l[i] = g[i];
  }
  for (int j = 0; j < 32; ++j) {
    int f = j * 256 + tid;
    int r = f >> 9, k = f & 511;
    int s0 = sym[(size_t)(b0 + r) * T_LEN];
    alpha[f] = (k == 0) ? logB[s0 * S_DIM] : -1e30f;
  }
  __syncthreads();

  const int mrow_off    = (lane >> 4) * 8;
  const int col_in_tile = lane & 15;
  const int bkoff       = (lane >> 4) * 16;
  const int akoff       = (lane >> 4) * 16;

  for (int t = 1; t < T_LEN; ++t) {
    if (tid < 16) symb[tid] = sym[(size_t)(b0 + tid) * T_LEN + t];
    {
      int r = tid & 15, c = tid >> 4;
      const float* ap = alpha + r * S_DIM + c * 32;
      float mx = ap[0];
      #pragma unroll 8
      for (int j = 1; j < 32; ++j) mx = fmaxf(mx, ap[j]);
      red[r * 16 + c] = mx;
    }
    __syncthreads();
    if (tid < 16) {
      float mx = red[tid * 16];
      #pragma unroll
      for (int c = 1; c < 16; ++c) mx = fmaxf(mx, red[tid * 16 + c]);
      mrow[tid] = mx;
    }
    __syncthreads();

    for (int j = 0; j < 32; ++j) {
      int f = j * 256 + tid;
      int r = f >> 9;
      P_lds[f] = (unsigned char)f32_to_fp8_e4m3(__expf(alpha[f] - mrow[r]));
    }
    __syncthreads();

    v16i pa[4];
    {
      const unsigned char* prow = P_lds + (lane & 15) * S_DIM;
      #pragma unroll
      for (int kt = 0; kt < 4; ++kt) {
        v16i a;
        #pragma unroll
        for (int h = 0; h < 2; ++h) {
          int base = kt * 128 + h * 64 + akoff;
          uint2 a0 = *(const uint2*)(prow + base);
          uint2 a1 = *(const uint2*)(prow + base + 8);
          uint2 a2 = *(const uint2*)(prow + base + 32);
          uint2 a3 = *(const uint2*)(prow + base + 40);
          a[8 * h + 0] = a0.x; a[8 * h + 1] = a0.y;
          a[8 * h + 2] = a1.x; a[8 * h + 3] = a1.y;
          a[8 * h + 4] = a2.x; a[8 * h + 5] = a2.y;
          a[8 * h + 6] = a3.x; a[8 * h + 7] = a3.y;
        }
        pa[kt] = a;
      }
    }
    #pragma unroll
    for (int nt = 0; nt < 4; ++nt) {
      int ntile = wave * 4 + nt;
      int n = ntile * 16 + col_in_tile;
      const unsigned char* arow = A_lds + n * S_DIM;
      v8f acc = {0.f, 0.f, 0.f, 0.f, 0.f, 0.f, 0.f, 0.f};
      #pragma unroll
      for (int kt = 0; kt < 4; ++kt) {
        v16i bf;
        #pragma unroll
        for (int q = 0; q < 4; ++q) {
          uint4 b = *(const uint4*)(arow + kt * 128 + q * 32 + bkoff);
          bf[4 * q + 0] = b.x; bf[4 * q + 1] = b.y;
          bf[4 * q + 2] = b.z; bf[4 * q + 3] = b.w;
        }
        acc = __builtin_amdgcn_wmma_f32_16x16x128_fp8_fp8(
            pa[kt], bf, (short)0, acc, false, false);
      }
      #pragma unroll
      for (int i = 0; i < 8; ++i) {
        int row = i + mrow_off;
        float R  = __logf(acc[i] * INV_A_SCALE + F_EPS) + mrow[row];
        float lE = logB[symb[row] * S_DIM + n];
        alpha[row * S_DIM + n] = lE + R;
      }
    }
    __syncthreads();
  }

  {
    int r = tid & 15, c = tid >> 4;
    const float* ap = alpha + r * S_DIM + c * 32;
    float mx = ap[0];
    for (int j = 1; j < 32; ++j) mx = fmaxf(mx, ap[j]);
    red[r * 16 + c] = mx;
  }
  __syncthreads();
  if (tid < 16) {
    float mx = red[tid * 16];
    for (int c = 1; c < 16; ++c) mx = fmaxf(mx, red[tid * 16 + c]);
    mrow[tid] = mx;
  }
  __syncthreads();
  {
    int r = tid & 15, c = tid >> 4;
    const float* ap = alpha + r * S_DIM + c * 32;
    float s = 0.f;
    for (int j = 0; j < 32; ++j) s += __expf(ap[j] - mrow[r]) + F_EPS;
    red[r * 16 + c] = s;
  }
  __syncthreads();
  if (tid < 16) {
    float s = 0.f;
    for (int c = 0; c < 16; ++c) s += red[tid * 16 + c];
    out_ll[b0 + tid] = __logf(s) + mrow[tid];
  }
  for (int j = 0; j < 32; ++j) {
    int f = j * 256 + tid;
    int r = f >> 9, k = f & 511;
    out_alpha[(size_t)(b0 + r) * S_DIM + k] = alpha[f];
  }
}

extern "C" void kernel_launch(void* const* d_in, const int* in_sizes, int n_in,
                              void* d_out, int out_size, void* d_ws, size_t ws_size,
                              hipStream_t stream) {
  (void)in_sizes; (void)n_in; (void)out_size; (void)ws_size;
  const float* x  = (const float*)d_in[0];
  const float* A  = (const float*)d_in[1];
  const float* Bm = (const float*)d_in[2];

  unsigned char* A8t  = (unsigned char*)d_ws;
  float*         logB = (float*)((char*)d_ws + 262144);
  unsigned char* sym  = (unsigned char*)d_ws + 262144 + 131072;

  float* out_alpha = (float*)d_out;
  float* out_ll    = out_alpha + B_SZ * S_DIM;

  hipLaunchKernelGGL(k_conv_A, dim3((S_DIM * S_DIM) / 256), dim3(256), 0, stream, A, A8t);
  hipLaunchKernelGGL(k_logB,   dim3((M_DIM * S_DIM) / 256), dim3(256), 0, stream, Bm, logB);
  hipLaunchKernelGGL(k_sym,    dim3((B_SZ * T_LEN) / 256),  dim3(256), 0, stream, x, sym);
  hipLaunchKernelGGL(k_forward, dim3(B_SZ / 16), dim3(256), LDS_TOTAL, stream,
                     A8t, logB, sym, out_alpha, out_ll);
}
